// DecayedLinearState_75917841924770
// MI455X (gfx1250) — matrix operand order repaired and checked by reading, not compiled, not run on hardware
//
#include <hip/hip_runtime.h>

// Problem constants (match reference)
#define Bq 16
#define Tq 8192
#define Dq 128
#define Cq 64
#define Nq (Tq / Cq)   // 128 chunks
#define JG 4           // column groups per batch (independent state slices)
#define JW (Dq / JG)   // 32 output/state columns per block
#define NTHREADS 256
#define NWAVES 8

typedef __attribute__((ext_vector_type(16))) __bf16 v16bf;
typedef __attribute__((ext_vector_type(8)))  float  v8f;

#define WMMA_BF16(acc, a, b)                                                  \
  (acc) = __builtin_amdgcn_wmma_f32_16x16x32_bf16(false, (a), false, (b),     \
                                                  (short)0, (acc), false, false)

struct SmemLayout {
  float S[Dq * JW];      // f32 master state slice, 128 x 32
  float attn[Cq * Cq];   // decayed attention tile, 64 x 64
  float dm[Cq * Cq];     // intra-chunk decay matrix lam^(i-j) * causal
  float z[Dq];           // f32 z state
  float tz[Cq];          // per-row total normalizer
  float dqv[Cq];         // lam^(c+1)
  float cdv[Cq];         // lam^(C-1-c)
  unsigned short Qb[Cq * Dq];  // bf16 Q chunk 64x128
  unsigned short Kb[Cq * Dq];  // bf16 K chunk 64x128
  unsigned short Vb[Cq * JW];  // bf16 V slice 64x32
};

// --- ISA fragment index helpers (wave32) ---
// 16-bit A 16x32, this code's contiguous k order: lanes 0-15 rows 0-15 K 0..15; lanes 16-31 K 16..31
__device__ __forceinline__ int a_koff(int e, int h) {
  return e + h * 16;
}

// A fragment from bf16 row-major tile
__device__ __forceinline__ v16bf ldA_bf(const unsigned short* src, int ld,
                                        int r0, int c0, int lane) {
  const __bf16* s = reinterpret_cast<const __bf16*>(src);
  int m = lane & 15, h = lane >> 4;
  v16bf a;
#pragma unroll
  for (int e = 0; e < 16; ++e)
    a[e] = s[(r0 + m) * ld + c0 + a_koff(e, h)];
  return a;
}

// A fragment from f32 row-major tile (convert to bf16 at load)
__device__ __forceinline__ v16bf ldA_f32(const float* s, int ld,
                                         int r0, int c0, int lane) {
  int m = lane & 15, h = lane >> 4;
  v16bf a;
#pragma unroll
  for (int e = 0; e < 16; ++e)
    a[e] = (__bf16)s[(r0 + m) * ld + c0 + a_koff(e, h)];
  return a;
}

// A fragment of (K^T * diag(cd)): A(i, c) = K[c][i] * cd[c]
__device__ __forceinline__ v16bf ldA_ktcd(const unsigned short* Ksrc,
                                          const float* cd, int i0, int c0,
                                          int lane) {
  const __bf16* s = reinterpret_cast<const __bf16*>(Ksrc);
  int m = lane & 15, h = lane >> 4;
  v16bf a;
#pragma unroll
  for (int e = 0; e < 16; ++e) {
    int c = c0 + a_koff(e, h);
    a[e] = (__bf16)((float)s[c * Dq + i0 + m] * cd[c]);
  }
  return a;
}

// B fragment 32x16: lane holds column n=lane&15; K 0-15 (lanes 0-15), 16-31 (lanes 16-31)
__device__ __forceinline__ v16bf ldB_bf(const unsigned short* src, int ld,
                                        int k0, int n0, int lane) {
  const __bf16* s = reinterpret_cast<const __bf16*>(src);
  int n = lane & 15, h = lane >> 4;
  v16bf b;
#pragma unroll
  for (int e = 0; e < 16; ++e)
    b[e] = s[(k0 + h * 16 + e) * ld + n0 + n];
  return b;
}

// B fragment of a transposed bf16 matrix: B(k, n) = src[n][k] (for Q.K^T)
__device__ __forceinline__ v16bf ldBT_bf(const unsigned short* src, int ld,
                                         int k0, int n0, int lane) {
  const __bf16* s = reinterpret_cast<const __bf16*>(src);
  int n = lane & 15, h = lane >> 4;
  v16bf b;
#pragma unroll
  for (int e = 0; e < 16; ++e)
    b[e] = s[(n0 + n) * ld + k0 + h * 16 + e];
  return b;
}

// B fragment from f32 matrix (state), convert to bf16 at load
__device__ __forceinline__ v16bf ldB_f32(const float* s, int ld, int k0,
                                         int n0, int lane) {
  int n = lane & 15, h = lane >> 4;
  v16bf b;
#pragma unroll
  for (int e = 0; e < 16; ++e)
    b[e] = (__bf16)s[(k0 + h * 16 + e) * ld + n0 + n];
  return b;
}

__global__ __launch_bounds__(NTHREADS)
void DecayedLinearState_kernel(const float* __restrict__ q,
                               const float* __restrict__ k,
                               const float* __restrict__ v,
                               const float* __restrict__ log_decay,
                               float* __restrict__ out) {
  extern __shared__ char smem_raw[];
  SmemLayout& sm = *reinterpret_cast<SmemLayout*>(smem_raw);

  const int jg   = blockIdx.x;   // column-slice group
  const int b    = blockIdx.y;   // batch
  const int tid  = threadIdx.x;
  const int wv   = tid >> 5;
  const int lane = tid & 31;

  const float lam    = 1.0f / (1.0f + __expf(-log_decay[0]));
  const float loglam = __logf(lam);
  const float lamC   = __expf(loglam * (float)Cq);

  // --- init decay tables, state, z ---
  for (int i = tid; i < Cq * Cq; i += NTHREADS) {
    int r = i >> 6, c = i & 63;
    sm.dm[i] = (r >= c) ? __expf(loglam * (float)(r - c)) : 0.0f;
  }
  for (int i = tid; i < Dq * JW; i += NTHREADS) sm.S[i] = 0.0f;
  if (tid < Dq) sm.z[tid] = 0.0f;
  if (tid < Cq) {
    sm.dqv[tid] = __expf(loglam * (float)(tid + 1));
    sm.cdv[tid] = __expf(loglam * (float)(Cq - 1 - tid));
  }
  __syncthreads();

  const float* qb = q + (size_t)b * Tq * Dq;
  const float* kb = k + (size_t)b * Tq * Dq;
  const float* vb = v + (size_t)b * Tq * Dq;

  for (int nc = 0; nc < Nq; ++nc) {
    const size_t base = (size_t)nc * Cq * Dq;

    // --- stage chunk: Q,K full (64x128) as bf16; V column slice (64x32) ---
    {
      __bf16* Q = reinterpret_cast<__bf16*>(sm.Qb);
      __bf16* K = reinterpret_cast<__bf16*>(sm.Kb);
      for (int i = tid; i < (Cq * Dq) / 4; i += NTHREADS) {
        float4 fq = ((const float4*)(qb + base))[i];
        float4 fk = ((const float4*)(kb + base))[i];
        int o = i * 4;
        Q[o] = (__bf16)fq.x; Q[o+1] = (__bf16)fq.y;
        Q[o+2] = (__bf16)fq.z; Q[o+3] = (__bf16)fq.w;
        K[o] = (__bf16)fk.x; K[o+1] = (__bf16)fk.y;
        K[o+2] = (__bf16)fk.z; K[o+3] = (__bf16)fk.w;
      }
      __bf16* V = reinterpret_cast<__bf16*>(sm.Vb);
      for (int i = tid; i < (Cq * JW) / 4; i += NTHREADS) {
        int row = i >> 3;            // 8 float4 per 32-col row
        int c4  = (i & 7) * 4;
        float4 fv = *(const float4*)(vb + base + (size_t)row * Dq + jg * JW + c4);
        int o = row * JW + c4;
        V[o] = (__bf16)fv.x; V[o+1] = (__bf16)fv.y;
        V[o+2] = (__bf16)fv.z; V[o+3] = (__bf16)fv.w;
      }
      // prefetch next chunk's Q/K (global_prefetch_b8)
      if (nc + 1 < Nq) {
        __builtin_prefetch(qb + base + Cq * Dq + tid * 32, 0, 1);
        __builtin_prefetch(kb + base + Cq * Dq + tid * 32, 0, 1);
      }
    }
    __syncthreads();

    // --- Phase A: attn = (Q.K^T) .* decay, f32 in LDS ---
#pragma unroll
    for (int ti = 0; ti < 2; ++ti) {
      int t = wv * 2 + ti;
      int tr = t >> 2, tc = t & 3;     // 4x4 tiles over 64x64
      v8f acc = {};
#pragma unroll
      for (int kk = 0; kk < Dq; kk += 32) {
        v16bf a  = ldA_bf(sm.Qb, Dq, tr * 16, kk, lane);
        v16bf bm = ldBT_bf(sm.Kb, Dq, kk, tc * 16, lane);
        WMMA_BF16(acc, a, bm);
      }
      int n = lane & 15, h = lane >> 4;
#pragma unroll
      for (int r = 0; r < 8; ++r) {
        int i_ = tr * 16 + r + h * 8, j_ = tc * 16 + n;
        sm.attn[i_ * Cq + j_] = acc[r] * sm.dm[i_ * Cq + j_];
      }
    }
    __syncthreads();

    // --- Phase A2: total normalizer tz[c] = max(rowsum(attn) + dq*(q.z), 1) ---
    if (tid < Cq) {
      float s1 = 0.0f;
      for (int j = 0; j < Cq; ++j) s1 += sm.attn[tid * Cq + j];
      float s2 = 0.0f;
      const __bf16* Q = reinterpret_cast<const __bf16*>(sm.Qb);
      for (int i = 0; i < Dq; ++i) s2 += (float)Q[tid * Dq + i] * sm.z[i];
      sm.tz[tid] = fmaxf(s1 + s2 * sm.dqv[tid], 1.0f);
    }
    __syncthreads();

    // --- Phase B: out = (dq .* (Q @ S) + attn @ V) / tz ---
    {
      int tr = wv >> 1, tc = wv & 1;   // 4x2 tiles over 64x32, one per wave
      v8f acc = {};
#pragma unroll
      for (int kk = 0; kk < Dq; kk += 32) {
        v16bf a  = ldA_bf(sm.Qb, Dq, tr * 16, kk, lane);
        v16bf bm = ldB_f32(sm.S, JW, kk, tc * 16, lane);
        WMMA_BF16(acc, a, bm);
      }
      int n = lane & 15, h = lane >> 4;
#pragma unroll
      for (int r = 0; r < 8; ++r) acc[r] *= sm.dqv[tr * 16 + r + h * 8];
#pragma unroll
      for (int kk = 0; kk < Cq; kk += 32) {
        v16bf a  = ldA_f32(sm.attn, Cq, tr * 16, kk, lane);
        v16bf bm = ldB_bf(sm.Vb, JW, kk, tc * 16, lane);
        WMMA_BF16(acc, a, bm);
      }
#pragma unroll
      for (int r = 0; r < 8; ++r) {
        int row = tr * 16 + r + h * 8;
        int col = tc * 16 + n;
        float val = acc[r] / sm.tz[row];
        out[((size_t)b * Tq + (size_t)nc * Cq + row) * Dq + jg * JW + col] = val;
      }
    }
    __syncthreads();

    // --- Phase C: S = lam^C * S + K^T diag(cd) V ; z = lam^C * z + K^T cd ---
#pragma unroll
    for (int ti = 0; ti < 2; ++ti) {
      int t = wv * 2 + ti;
      int tr = t >> 1, tc = t & 1;     // 8x2 tiles over 128x32
      v8f acc = {};
#pragma unroll
      for (int kk = 0; kk < Cq; kk += 32) {
        v16bf a  = ldA_ktcd(sm.Kb, sm.cdv, tr * 16, kk, lane);
        v16bf bm = ldB_bf(sm.Vb, JW, kk, tc * 16, lane);
        WMMA_BF16(acc, a, bm);
      }
      int n = lane & 15, h = lane >> 4;
#pragma unroll
      for (int r = 0; r < 8; ++r) {
        int i_ = tr * 16 + r + h * 8, j_ = tc * 16 + n;
        sm.S[i_ * JW + j_] = lamC * sm.S[i_ * JW + j_] + acc[r];
      }
    }
    if (tid < Dq) {
      const __bf16* K = reinterpret_cast<const __bf16*>(sm.Kb);
      float s = 0.0f;
      for (int c = 0; c < Cq; ++c) s += (float)K[c * Dq + tid] * sm.cdv[c];
      sm.z[tid] = lamC * sm.z[tid] + s;
    }
    __syncthreads();
  }

  // --- final state and z ---
  const size_t stateBase = (size_t)Bq * Tq * Dq;
  for (int i = tid; i < Dq * JW; i += NTHREADS) {
    int r = i / JW, c = i % JW;
    out[stateBase + (size_t)b * Dq * Dq + (size_t)r * Dq + jg * JW + c] = sm.S[i];
  }
  if (jg == 0 && tid < Dq)
    out[stateBase + (size_t)Bq * Dq * Dq + (size_t)b * Dq + tid] = sm.z[tid];
}

extern "C" void kernel_launch(void* const* d_in, const int* in_sizes, int n_in,
                              void* d_out, int out_size, void* d_ws, size_t ws_size,
                              hipStream_t stream) {
  (void)in_sizes; (void)n_in; (void)d_ws; (void)ws_size; (void)out_size;
  const float* q  = (const float*)d_in[0];
  const float* k  = (const float*)d_in[1];
  const float* v  = (const float*)d_in[2];
  const float* ld = (const float*)d_in[3];
  float* out = (float*)d_out;

  dim3 grid(JG, Bq);          // 4 column slices x 16 batches = 64 WGs
  dim3 block(NTHREADS);       // 8 wave32
  size_t shmem = sizeof(SmemLayout);  // ~85 KB, fits 320 KB WGP LDS
  DecayedLinearState_kernel<<<grid, block, shmem, stream>>>(q, k, v, ld, out);
}
